// Attention_26680336843301
// MI455X (gfx1250) — compile-verified, matrix operand order checked by reading, not run on hardware
//
#include <hip/hip_runtime.h>

// T5 encoder self-attention, B=2, L=2048, D=1024, H=16, hd=64, fp32 I/O.
// Matmuls: v_wmma_f32_16x16x32_f16.  Staging: TDM tensor_load_to_lds
// (TENSORcnt, HW LDS padding) in the GEMMs; global_load_async_to_lds
// (ASYNCcnt) in the attention kernel.

#define NHEADS 16
#define BB 2
#define LL 2048
#define DD 1024
#define HDIM 64
#define LUTW 4095   // relpos in [-2047, 2047]

typedef _Float16 h16;
typedef __attribute__((ext_vector_type(16))) _Float16 v16h;
typedef __attribute__((ext_vector_type(8)))  _Float16 v8h;
typedef __attribute__((ext_vector_type(4)))  _Float16 v4h;
typedef __attribute__((ext_vector_type(8)))  float    v8f;
typedef __attribute__((ext_vector_type(4)))  int      v4i;
typedef __attribute__((ext_vector_type(4)))  unsigned int v4u;
typedef __attribute__((ext_vector_type(8)))  int      v8i;

typedef __attribute__((address_space(1))) v4i* as1_v4i;
typedef __attribute__((address_space(3))) v4i* as3_v4i;

#if __has_builtin(__builtin_amdgcn_tensor_load_to_lds) && \
    __has_builtin(__builtin_amdgcn_s_wait_tensorcnt)
#define USE_TDM 1
#else
#define USE_TDM 0
#endif

// ---------------------------------------------------------------------------
// CDNA5 async global->LDS copy (ASYNCcnt) + waits.
// ---------------------------------------------------------------------------
__device__ inline void async_cp16(void* lds, const void* g) {
#if __has_builtin(__builtin_amdgcn_global_load_async_to_lds_b128)
  __builtin_amdgcn_global_load_async_to_lds_b128(
      (as1_v4i)(__SIZE_TYPE__)g,
      (as3_v4i)(unsigned)(__SIZE_TYPE__)lds,
      0, 0);
#else
  asm volatile("global_load_async_to_lds_b128 %0, %1, off"
               :: "v"((unsigned)(__SIZE_TYPE__)lds),
                  "v"((unsigned long long)(__SIZE_TYPE__)g)
               : "memory");
#endif
}

__device__ inline void wait_async0() {
#if __has_builtin(__builtin_amdgcn_s_wait_asynccnt)
  __builtin_amdgcn_s_wait_asynccnt(0);
#else
  asm volatile("s_wait_asynccnt 0x0" ::: "memory");
#endif
}
__device__ inline void wait_async3() {
#if __has_builtin(__builtin_amdgcn_s_wait_asynccnt)
  __builtin_amdgcn_s_wait_asynccnt(3);
#else
  asm volatile("s_wait_asynccnt 0x3" ::: "memory");
#endif
}

// ---------------------------------------------------------------------------
// Tensor Data Mover: one 2D tile -> LDS, with HW row padding so LDS rows are
// (tileW*2 + 16) bytes. Descriptor per ISA ch.8 (D# groups 0/1).
// tileW in f16 elements, strideElems = global row stride in elements.
// padIntCode: pad after 2^(code+1) DWORDs; padAmtCode: pad (code+1) DWORDs.
// ---------------------------------------------------------------------------
#if USE_TDM
__device__ inline void tdm_load_2d(unsigned lds_addr, const void* gaddr,
                                   unsigned tileW, unsigned tileH,
                                   unsigned strideElems,
                                   unsigned padIntCode, unsigned padAmtCode) {
  unsigned long long ga = (unsigned long long)(__SIZE_TYPE__)gaddr;
  v4u g0;
  g0[0] = 1u;                                        // count=1, user D#
  g0[1] = lds_addr;                                  // LDS byte address
  g0[2] = (unsigned)(ga & 0xFFFFFFFFu);              // global_addr[31:0]
  g0[3] = (unsigned)((ga >> 32) & 0x1FFFFFFu)        // global_addr[56:32]
          | (2u << 30);                              // type=2 ("image")
  v8i g1;
  g1[0] = (int)((1u << 16)                           // data_size = 2 bytes
                | (1u << 20)                         // pad_enable
                | (padIntCode << 22) | (padAmtCode << 25));
  g1[1] = (int)((tileW & 0xFFFFu) << 16);            // tensor_dim0 = tileW
  g1[2] = (int)((tileH & 0xFFFFu) << 16);            // tensor_dim1 = tileH
  g1[3] = (int)((tileW & 0xFFFFu) << 16);            // tile_dim0
  g1[4] = (int)(tileH & 0xFFFFu);                    // tile_dim1 (tile_dim2=0)
  g1[5] = (int)strideElems;                          // tensor_dim0_stride lo32
  g1[6] = 0;                                         // stride hi / dim1_stride
  g1[7] = 0;
  v4i z4 = {0, 0, 0, 0};
  v8i z8 = {0, 0, 0, 0, 0, 0, 0, 0};
  __builtin_amdgcn_tensor_load_to_lds(g0, g1, z4, z4, z8, 0);
}
__device__ inline void wait_tensor0() { __builtin_amdgcn_s_wait_tensorcnt(0); }
__device__ inline void wait_tensor2() { __builtin_amdgcn_s_wait_tensorcnt(2); }
#endif

// ---------------------------------------------------------------------------
// WMMA fragment loader (ISA 16-bit 16x32 layout: lanes 0-15 row=lane hold
// K=0..7 then 16..23; lanes 16-31 hold K=8..15 then 24..31).
// ---------------------------------------------------------------------------
__device__ inline v16h frag_ld(const h16* __restrict__ p, int ld) {
  const int lane = threadIdx.x & 31;
  const int row  = lane & 15;
  const int k0   = (lane & 16) ? 8 : 0;
  const h16* bp  = p + (__SIZE_TYPE__)row * ld + k0;
  v8h lo = *(const v8h*)(bp);
  v8h hi = *(const v8h*)(bp + 16);
  v16h r;
#pragma unroll
  for (int i = 0; i < 8; ++i) { r[i] = lo[i]; r[i + 8] = hi[i]; }
  return r;
}

__device__ inline v8f wmma_f16(v16h a, v16h b, v8f c) {
  return __builtin_amdgcn_wmma_f32_16x16x32_f16(false, a, false, b,
                                                (short)0, c, false, false);
}

// ---------------------------------------------------------------------------
// fp32 -> f16 pre-conversion (once; enables pure-DMA tile staging later)
// ---------------------------------------------------------------------------
__global__ void cvt_f32_f16(const float* __restrict__ in,
                            h16* __restrict__ out, int n4) {
  int i = blockIdx.x * 256 + threadIdx.x;
  if (i >= n4) return;
  float4 v = ((const float4*)in)[i];
  v4h o;
  o[0] = (h16)v.x; o[1] = (h16)v.y; o[2] = (h16)v.z; o[3] = (h16)v.w;
  ((v4h*)out)[i] = o;
}

// ---------------------------------------------------------------------------
// T5 relative-position bias LUT: LUT[h*4095 + (rel+2047)]
// ---------------------------------------------------------------------------
__global__ void bias_lut_kernel(const float* __restrict__ table,
                                float* __restrict__ LUT) {
  int i = blockIdx.x * 256 + threadIdx.x;
  if (i >= NHEADS * LUTW) return;
  int h   = i / LUTW;
  int rel = (i % LUTW) - 2047;          // mem - ctx
  int bucket = (rel > 0) ? 16 : 0;      // nb = 16
  int rp = (rel < 0) ? -rel : rel;
  if (rp < 8) {
    bucket += rp;
  } else {
    int large = 8 + (int)(__logf((float)rp * 0.125f) *
                          (8.0f / __logf(16.0f)));
    bucket += (large > 15) ? 15 : large;
  }
  LUT[i] = table[bucket * NHEADS + h];
}

// ---------------------------------------------------------------------------
// stream position_bias [1,H,L,L] fp32 to output (268 MB; BW-bound)
// ---------------------------------------------------------------------------
__global__ void bias_out_kernel(const float* __restrict__ LUT,
                                float4* __restrict__ outB) {
  __SIZE_TYPE__ i4 = (__SIZE_TYPE__)blockIdx.x * 256 + threadIdx.x;
  const __SIZE_TYPE__ total = (__SIZE_TYPE__)NHEADS * LL * (LL / 4);
  if (i4 >= total) return;
  int k4 = (int)(i4 % (LL / 4));
  __SIZE_TYPE__ t = i4 / (LL / 4);
  int q = (int)(t % LL);
  int h = (int)(t / LL);
  const float* lp = LUT + h * LUTW + 2047 + (k4 * 4 - q);
  float4 v; v.x = lp[0]; v.y = lp[1]; v.z = lp[2]; v.w = lp[3];
  outB[i4] = v;
}

// ---------------------------------------------------------------------------
// Y[M,N] = X[M,K] @ W[N,K]^T, f16 in, f32 accum. Block tile 128x64, BK=32.
// Double-buffered LDS tiles filled by the Tensor Data Mover (one descriptor
// per tile, HW-padded rows: 64B data + 16B pad = 80B = LDA 40 halves,
// conflict-free for ds_load_b128 fragment reads).
// ---------------------------------------------------------------------------
template <typename TOUT>
__global__ __launch_bounds__(256) void gemm_f16(const h16* __restrict__ X,
                                                const h16* __restrict__ W,
                                                TOUT* __restrict__ Y,
                                                int M, int N, int K) {
  const int LDA = 40;
  __shared__ h16 sX[2][128 * 40];
  __shared__ h16 sW[2][64 * 40];

  const int bm = blockIdx.y * 128;
  const int bn = blockIdx.x * 64;
  const int t = threadIdx.x;
  const int wave = t >> 5;
  const int wm = (wave & 3) * 32;
  const int wn = (wave >> 2) * 32;

#if !USE_TDM
  // async fallback staging: sX = 512 b128-chunks (2/thread), sW = 256
  const int xr0 = t >> 2,          xs0 = (t & 3) * 8;
  const int xr1 = (t + 256) >> 2,  xs1 = ((t + 256) & 3) * 8;
  const int wr  = t >> 2,          wsb = (t & 3) * 8;
#endif

  v8f c00 = {}, c01 = {}, c10 = {}, c11 = {};

  auto issue = [&](int kk, int buf) {
#if USE_TDM
    if (wave == 0) {
      // rows: 32 halves = 16 DW -> pad_interval code 3; pad 4 DW -> code 3
      tdm_load_2d((unsigned)(__SIZE_TYPE__)&sX[buf][0],
                  X + (__SIZE_TYPE__)bm * K + kk, 32, 128, (unsigned)K, 3, 3);
      tdm_load_2d((unsigned)(__SIZE_TYPE__)&sW[buf][0],
                  W + (__SIZE_TYPE__)bn * K + kk, 32, 64, (unsigned)K, 3, 3);
    }
#else
    async_cp16(&sX[buf][xr0 * LDA + xs0],
               X + (__SIZE_TYPE__)(bm + xr0) * K + kk + xs0);
    async_cp16(&sX[buf][xr1 * LDA + xs1],
               X + (__SIZE_TYPE__)(bm + xr1) * K + kk + xs1);
    async_cp16(&sW[buf][wr * LDA + wsb],
               W + (__SIZE_TYPE__)(bn + wr) * K + kk + wsb);
#endif
  };

  issue(0, 0);
  int cur = 0;
  for (int kk = 0; kk < K; kk += 32) {
    bool more = (kk + 32) < K;
    if (more) issue(kk + 32, cur ^ 1);      // prefetch next tile
#if USE_TDM
    if (more) wait_tensor2(); else wait_tensor0();  // tensor ops in-order
#else
    if (more) wait_async3(); else wait_async0();
#endif
    __syncthreads();

    v16h a0 = frag_ld(&sX[cur][(wm + 0)  * LDA], LDA);
    v16h a1 = frag_ld(&sX[cur][(wm + 16) * LDA], LDA);
    v16h b0 = frag_ld(&sW[cur][(wn + 0)  * LDA], LDA);
    v16h b1 = frag_ld(&sW[cur][(wn + 16) * LDA], LDA);
    c00 = wmma_f16(a0, b0, c00);
    c01 = wmma_f16(a0, b1, c01);
    c10 = wmma_f16(a1, b0, c10);
    c11 = wmma_f16(a1, b1, c11);

    __syncthreads();                         // reads done before next overwrite
    cur ^= 1;
  }

  const int lane = t & 31;
  const int col = lane & 15;
  const int rhi = (lane & 16) ? 8 : 0;
  v8f* cs[4] = { &c00, &c01, &c10, &c11 };
#pragma unroll
  for (int q = 0; q < 4; ++q) {
    int mi = q >> 1, ni = q & 1;
#pragma unroll
    for (int j = 0; j < 8; ++j) {
      int r  = bm + wm + mi * 16 + j + rhi;
      int cc = bn + wn + ni * 16 + col;
      Y[(__SIZE_TYPE__)r * N + cc] = (TOUT)((*cs[q])[j]);
    }
  }
}

// ---------------------------------------------------------------------------
// fused flash attention with T5 bias. Block = one (b,h) x 128 query rows;
// 8 waves x 16 queries; 64-key chunks. K tile staged by async-to-LDS DMA.
// ---------------------------------------------------------------------------
__global__ __launch_bounds__(256) void attn_kernel(const h16* __restrict__ Q,
                                                   const h16* __restrict__ Km,
                                                   const h16* __restrict__ V,
                                                   const float* __restrict__ LUT,
                                                   h16* __restrict__ Ctx) {
  const int LDK = 72;                     // 144B rows, 16B aligned chunks
  __shared__ h16 sK [64 * 72];            // keys x hd (row-major)
  __shared__ h16 sVt[64 * 72];            // hd x keys (V transposed)
  __shared__ h16 sP [8 * 16 * 72];        // per-wave P scratch

  const int bh = blockIdx.y;
  const int b = bh >> 4, h = bh & 15;
  const int q0 = blockIdx.x * 128;
  const int t = threadIdx.x, lane = t & 31, wave = t >> 5;
  const int qw = q0 + wave * 16;
  const int col = lane & 15;
  const int rhi = (lane & 16) ? 8 : 0;

  const __SIZE_TYPE__ headOff = ((__SIZE_TYPE__)b * LL) * DD +
                                (__SIZE_TYPE__)h * HDIM;

  // K staging: 512 b128-chunks -> 2/thread ; V staged + transposed manually
  const int kr0 = t >> 3,          ks0 = (t & 7) * 8;
  const int kr1 = (t + 256) >> 3,  ks1 = ((t + 256) & 7) * 8;

  v16h aq0 = frag_ld(Q + headOff + (__SIZE_TYPE__)qw * DD, DD);
  v16h aq1 = frag_ld(Q + headOff + (__SIZE_TYPE__)qw * DD + 32, DD);

  v8f o0 = {}, o1 = {}, o2 = {}, o3 = {};
  float m[8], l[8];
#pragma unroll
  for (int j = 0; j < 8; ++j) { m[j] = -1e30f; l[j] = 0.f; }

  for (int kk0 = 0; kk0 < LL; kk0 += 64) {
    // K tile: async DMA straight into LDS
    async_cp16(&sK[kr0 * LDK + ks0],
               Km + headOff + (__SIZE_TYPE__)(kk0 + kr0) * DD + ks0);
    async_cp16(&sK[kr1 * LDK + ks1],
               Km + headOff + (__SIZE_TYPE__)(kk0 + kr1) * DD + ks1);
    // V tile: vector load + transposed LDS store
#pragma unroll
    for (int i = 0; i < 2; ++i) {
      int r = (i == 0) ? kr0 : kr1, sb = (i == 0) ? ks0 : ks1;
      v8h vv = *(const v8h*)(V + headOff + (__SIZE_TYPE__)(kk0 + r) * DD + sb);
#pragma unroll
      for (int e = 0; e < 8; ++e) sVt[(sb + e) * LDK + r] = vv[e];
    }
    wait_async0();
    __syncthreads();

    // S = Q @ K^T + bias   (4 N-tiles of 16 keys)
    v8f s[4];
#pragma unroll
    for (int n = 0; n < 4; ++n) {
      v16h bk0 = frag_ld(sK + n * 16 * LDK, LDK);
      v16h bk1 = frag_ld(sK + n * 16 * LDK + 32, LDK);
      v8f acc = {};
      acc = wmma_f16(aq0, bk0, acc);
      acc = wmma_f16(aq1, bk1, acc);
      // relpos = (kk0+n*16+col) - (qw+j+rhi) = base - j
      int base = h * LUTW + 2047 + (kk0 + n * 16 + col) - (qw + rhi);
      const float* lp = LUT + base;
#pragma unroll
      for (int j = 0; j < 8; ++j) acc[j] += lp[-j];
      s[n] = acc;
    }

    // online softmax (rows of S and O fragments share j/rhi mapping)
    float scale[8];
#pragma unroll
    for (int j = 0; j < 8; ++j) {
      float v = fmaxf(fmaxf(s[0][j], s[1][j]), fmaxf(s[2][j], s[3][j]));
      v = fmaxf(v, __shfl_xor(v, 1, 32));
      v = fmaxf(v, __shfl_xor(v, 2, 32));
      v = fmaxf(v, __shfl_xor(v, 4, 32));
      v = fmaxf(v, __shfl_xor(v, 8, 32));
      float mn = fmaxf(m[j], v);
      scale[j] = __expf(m[j] - mn);
      m[j] = mn;
    }

    h16* pw = sP + wave * 16 * LDK;
    float rs[8] = {0, 0, 0, 0, 0, 0, 0, 0};
#pragma unroll
    for (int n = 0; n < 4; ++n) {
#pragma unroll
      for (int j = 0; j < 8; ++j) {
        float p = __expf(s[n][j] - m[j]);
        rs[j] += p;
        pw[(j + rhi) * LDK + n * 16 + col] = (h16)p;
      }
    }
#pragma unroll
    for (int j = 0; j < 8; ++j) {
      float v = rs[j];
      v += __shfl_xor(v, 1, 32);
      v += __shfl_xor(v, 2, 32);
      v += __shfl_xor(v, 4, 32);
      v += __shfl_xor(v, 8, 32);
      l[j] = l[j] * scale[j] + v;
      o0[j] *= scale[j]; o1[j] *= scale[j];
      o2[j] *= scale[j]; o3[j] *= scale[j];
    }

    // O += P @ V  (P via per-wave LDS round-trip into A-fragment layout)
    v16h ap0 = frag_ld(pw, LDK);
    v16h ap1 = frag_ld(pw + 32, LDK);
    {
      v16h bv0 = frag_ld(sVt + 0  * LDK, LDK);
      v16h bv1 = frag_ld(sVt + 0  * LDK + 32, LDK);
      o0 = wmma_f16(ap0, bv0, o0); o0 = wmma_f16(ap1, bv1, o0);
      bv0 = frag_ld(sVt + 16 * LDK, LDK);
      bv1 = frag_ld(sVt + 16 * LDK + 32, LDK);
      o1 = wmma_f16(ap0, bv0, o1); o1 = wmma_f16(ap1, bv1, o1);
      bv0 = frag_ld(sVt + 32 * LDK, LDK);
      bv1 = frag_ld(sVt + 32 * LDK + 32, LDK);
      o2 = wmma_f16(ap0, bv0, o2); o2 = wmma_f16(ap1, bv1, o2);
      bv0 = frag_ld(sVt + 48 * LDK, LDK);
      bv1 = frag_ld(sVt + 48 * LDK + 32, LDK);
      o3 = wmma_f16(ap0, bv0, o3); o3 = wmma_f16(ap1, bv1, o3);
    }
    __syncthreads();
  }

  h16* Cb = Ctx + headOff;
  v8f* os[4] = { &o0, &o1, &o2, &o3 };
#pragma unroll
  for (int d = 0; d < 4; ++d)
#pragma unroll
    for (int j = 0; j < 8; ++j) {
      float v = (*os[d])[j] / l[j];
      Cb[(__SIZE_TYPE__)(qw + j + rhi) * DD + d * 16 + col] = (h16)v;
    }
}

// ---------------------------------------------------------------------------
extern "C" void kernel_launch(void* const* d_in, const int* in_sizes, int n_in,
                              void* d_out, int out_size, void* d_ws,
                              size_t ws_size, hipStream_t stream) {
  (void)in_sizes; (void)n_in; (void)out_size; (void)ws_size;
  const float* hs = (const float*)d_in[0];
  const float* Wq = (const float*)d_in[1];
  const float* Wk = (const float*)d_in[2];
  const float* Wv = (const float*)d_in[3];
  const float* Wo = (const float*)d_in[4];
  const float* bt = (const float*)d_in[5];

  float* out   = (float*)d_out;                       // [B,L,D]
  float* pbias = out + (size_t)BB * LL * DD;          // [1,H,L,L]

  char* ws = (char*)d_ws;
  const size_t fsz = (size_t)BB * LL * DD;            // 4M elements
  const size_t wsz = (size_t)DD * DD;                 // 1M elements
  h16* Qf  = (h16*)ws; ws += fsz * 2;
  h16* Kf  = (h16*)ws; ws += fsz * 2;
  h16* Vf  = (h16*)ws; ws += fsz * 2;
  h16* Cf  = (h16*)ws; ws += fsz * 2;
  h16* hsF = (h16*)ws; ws += fsz * 2;
  h16* WqF = (h16*)ws; ws += wsz * 2;
  h16* WkF = (h16*)ws; ws += wsz * 2;
  h16* WvF = (h16*)ws; ws += wsz * 2;
  h16* WoF = (h16*)ws; ws += wsz * 2;
  float* LUT = (float*)ws;                            // 16*4095 floats

  // 0) one-shot fp32 -> f16 conversions (enables pure-DMA tile staging)
  cvt_f32_f16<<<(unsigned)(fsz / 4 / 256), 256, 0, stream>>>(hs, hsF,
                                                             (int)(fsz / 4));
  cvt_f32_f16<<<(unsigned)(wsz / 4 / 256), 256, 0, stream>>>(Wq, WqF,
                                                             (int)(wsz / 4));
  cvt_f32_f16<<<(unsigned)(wsz / 4 / 256), 256, 0, stream>>>(Wk, WkF,
                                                             (int)(wsz / 4));
  cvt_f32_f16<<<(unsigned)(wsz / 4 / 256), 256, 0, stream>>>(Wv, WvF,
                                                             (int)(wsz / 4));
  cvt_f32_f16<<<(unsigned)(wsz / 4 / 256), 256, 0, stream>>>(Wo, WoF,
                                                             (int)(wsz / 4));

  // 1) bias LUT
  bias_lut_kernel<<<(NHEADS * LUTW + 255) / 256, 256, 0, stream>>>(bt, LUT);

  // 2) position_bias output stream (268 MB, BW-bound)
  {
    size_t total = (size_t)NHEADS * LL * (LL / 4);
    bias_out_kernel<<<(unsigned)((total + 255) / 256), 256, 0, stream>>>(
        LUT, (float4*)pbias);
  }

  // 3) Q/K/V projections (f16 WMMA, TDM double-buffered tiles)
  dim3 gg(DD / 64, (BB * LL) / 128);
  gemm_f16<h16><<<gg, 256, 0, stream>>>(hsF, WqF, Qf, BB * LL, DD, DD);
  gemm_f16<h16><<<gg, 256, 0, stream>>>(hsF, WkF, Kf, BB * LL, DD, DD);
  gemm_f16<h16><<<gg, 256, 0, stream>>>(hsF, WvF, Vf, BB * LL, DD, DD);

  // 4) fused flash attention with T5 bias
  attn_kernel<<<dim3(LL / 128, BB * NHEADS), 256, 0, stream>>>(Qf, Kf, Vf,
                                                               LUT, Cf);

  // 5) output projection -> fp32 d_out
  gemm_f16<float><<<gg, 256, 0, stream>>>(Cf, WoF, out, BB * LL, DD, DD);
}
